// SelfAttentionHead_24120536335172
// MI455X (gfx1250) — matrix operand order repaired and checked by reading, not compiled, not run on hardware
//
#include <hip/hip_runtime.h>
#include <hip/hip_bf16.h>

// ---------------------------------------------------------------------------
// Self-attention head, B=4 T=4096 C=1024 HS=64, causal, scale = C^-0.5.
// bf16 WMMA everywhere (v_wmma_f32_16x16x32_bf16), flash-attention online
// softmax, x read once (fused QKV), V stored transposed so every WMMA
// B-fragment is a contiguous 32-byte load.
// ---------------------------------------------------------------------------

#define B_  4
#define T_  4096
#define C_  1024
#define HS_ 64

typedef __attribute__((ext_vector_type(16))) __bf16 v16bf;
typedef __attribute__((ext_vector_type(8)))  __bf16 v8bf;
typedef __attribute__((ext_vector_type(8)))  float  v8f;
typedef __attribute__((ext_vector_type(4)))  float  f4;

union AFrag { v16bf v; v8bf h[2]; };

// Native bf16 convert (v_cvt_pk_bf16_f32 on gfx1250).
static __device__ __forceinline__ unsigned short f2bf(float f) {
    union { __bf16 h; unsigned short u; } c;
    c.h = (__bf16)f;
    return c.u;
}

static __device__ __forceinline__ v8f wmma_bf16(v16bf a, v16bf b, v8f c) {
    return __builtin_amdgcn_wmma_f32_16x16x32_bf16(
        /*neg_a=*/false, a, /*neg_b=*/false, b,
        /*c_mod=*/(short)0, c, /*reuse_a=*/false, /*reuse_b=*/false);
}

// ---------------------------------------------------------------------------
// Kernel 0: transpose + convert weights fp32[C,64] -> bf16[64,C].
// Wt is one contiguous [12][16][C] tile array: tiles 0..3 = K, 4..7 = Q
// (scale 1/32 folded in), 8..11 = V.
// ---------------------------------------------------------------------------
__global__ void wt_convert_kernel(const float* __restrict__ Wk,
                                  const float* __restrict__ Wq,
                                  const float* __restrict__ Wv,
                                  unsigned short* __restrict__ Wt) {
    int idx = blockIdx.x * blockDim.x + threadIdx.x;     // over 64*1024
    if (idx >= HS_ * C_) return;
    int h = idx >> 10;          // head feature
    int c = idx & (C_ - 1);     // embedding channel
    Wt[idx]                     = f2bf(Wk[c * HS_ + h]);
    Wt[idx + HS_ * C_]          = f2bf(Wq[c * HS_ + h] * 0.03125f); // 1/sqrt(1024)
    Wt[idx + 2 * HS_ * C_]      = f2bf(Wv[c * HS_ + h]);
}

// ---------------------------------------------------------------------------
// Kernel 1: fused QKV projection, register-pressure balanced.
// Grid (T/64, B), block 256 (8 waves). Waves pair up: waves 2p,2p+1 share the
// same 16 rows of x (L2-resident, duplicate read is cheap); parity selects 6
// of the 12 contiguous (weight x column-subtile) output tiles. Per wave: 6
// accumulators (48 VGPRs) + 6 B-fragments (48 VGPRs) + A (8). B-fragment
// addresses are pure offsets from ONE base pointer (keeps global_load_b128
// inference; flat_load would cost DScnt too), so the 6 loads issue as one
// immediate-offset clause and WMMAs retire on partial loadcnt waits.
// Outputs: q,k bf16 row-major [B,T,64]; v bf16 transposed [B,64,T].
// ---------------------------------------------------------------------------
__global__ void qkv_proj_kernel(const float* __restrict__ x,
                                const unsigned short* __restrict__ Wt,
                                unsigned short* __restrict__ qbuf,
                                unsigned short* __restrict__ kbuf,
                                unsigned short* __restrict__ vTbuf) {
    const int lane = threadIdx.x & 31;
    const int wave = threadIdx.x >> 5;   // 0..7
    const int pair = wave >> 1;          // row-group 0..3
    const int par  = wave & 1;           // column split
    const int lrow = lane & 15;
    const int lhi  = lane >> 4;
    const int b    = blockIdx.y;
    const int rowbase = blockIdx.x * 64 + pair * 16;

    const float* xA = x + ((size_t)b * T_ + rowbase + lrow) * C_;

    // Single base for this wave's 6 tiles: tile t = par*6 + ci.
    const unsigned short* wb =
        Wt + (size_t)par * 6 * 16 * C_ + (size_t)lrow * C_ + lhi * 16;

    v8f acc[6];
    #pragma unroll
    for (int i = 0; i < 6; ++i) acc[i] = (v8f)(0.0f);

    for (int k0 = 0; k0 < C_; k0 += 32) {
        // ---- issue all 6 B-fragment loads (one clause, immediate offsets) ----
        v16bf bfr[6];
        #pragma unroll
        for (int ci = 0; ci < 6; ++ci)
            bfr[ci] = *(const v16bf*)(wb + (size_t)ci * 16 * C_ + k0);

        // ---- A fragment: 16x32 fp32 tile of x -> bf16 (contiguous k = 16*lhi + s, same order as B) ----
        const float* p0 = xA + k0 + lhi * 16;       // elements 0..7
        const float* p1 = p0 + 8;                   // elements 8..15
        f4 x0 = *(const f4*)(p0);
        f4 x1 = *(const f4*)(p0 + 4);
        f4 x2 = *(const f4*)(p1);
        f4 x3 = *(const f4*)(p1 + 4);
        AFrag a;
        #pragma unroll
        for (int j = 0; j < 4; ++j) {
            a.v[j]      = (__bf16)x0[j];
            a.v[4 + j]  = (__bf16)x1[j];
            a.v[8 + j]  = (__bf16)x2[j];
            a.v[12 + j] = (__bf16)x3[j];
        }

        // ---- 6 chained WMMAs ----
        #pragma unroll
        for (int ci = 0; ci < 6; ++ci)
            acc[ci] = wmma_bf16(a.v, bfr[ci], acc[ci]);
    }

    // Epilogue: C-layout lane l, reg r -> row (lhi*8+r), col (lrow).
    #pragma unroll
    for (int ci = 0; ci < 6; ++ci) {
        const int t = par * 6 + ci;           // global tile 0..11
        #pragma unroll
        for (int r = 0; r < 8; ++r) {
            const int rowg = rowbase + lhi * 8 + r;
            const unsigned short v = f2bf(acc[ci][r]);
            if (t < 4) {                      // K tiles 0..3 (row-major)
                kbuf[((size_t)b * T_ + rowg) * HS_ + t * 16 + lrow] = v;
            } else if (t < 8) {               // Q tiles 0..3 (row-major)
                qbuf[((size_t)b * T_ + rowg) * HS_ + (t - 4) * 16 + lrow] = v;
            } else {                          // V tiles 0..3 (transposed)
                vTbuf[((size_t)b * HS_ + (t - 8) * 16 + lrow) * T_ + rowg] = v;
            }
        }
    }
}

// ---------------------------------------------------------------------------
// Kernel 2: causal flash attention, 64-key tiles.
// Grid (T/64, B), block 128 (4 independent waves). Wave owns 16 queries.
// Per tile: 8 S-WMMAs, one shuffle-reduce chain per row covering all 64 keys,
// V fragments issued before the softmax VALU so their latency hides, then
// 8 P*V WMMAs. K/V are L2-resident (6 MB total, 192 MB L2).
// ---------------------------------------------------------------------------
__global__ void flash_attn_kernel(const unsigned short* __restrict__ qbuf,
                                  const unsigned short* __restrict__ kbuf,
                                  const unsigned short* __restrict__ vTbuf,
                                  float* __restrict__ out) {
    __shared__ unsigned short pLds[4][16 * 64];   // per-wave P staging tile

    const int lane = threadIdx.x & 31;
    const int wave = threadIdx.x >> 5;
    const int lrow = lane & 15;
    const int lhi  = lane >> 4;
    const int b    = blockIdx.y;
    const int qbase = blockIdx.x * 64 + wave * 16;

    const float L2E = 1.4426950408889634f;

    // Q tile in A-layout, loaded once: 2 fragments over the 64 features.
    const unsigned short* qrow = qbuf + ((size_t)b * T_ + qbase + lrow) * HS_;
    AFrag aq[2];
    #pragma unroll
    for (int kk = 0; kk < 2; ++kk) {
        aq[kk].h[0] = *(const v8bf*)(qrow + kk * 32 + lhi * 16);
        aq[kk].h[1] = *(const v8bf*)(qrow + kk * 32 + lhi * 16 + 8);
    }

    v8f O[4];
    #pragma unroll
    for (int t = 0; t < 4; ++t) O[t] = (v8f)(0.0f);
    float m[8], lsum[8];
    #pragma unroll
    for (int r = 0; r < 8; ++r) { m[r] = -1e30f; lsum[r] = 0.0f; }

    const int ntiles = (qbase + 16 + 63) >> 6;    // keys 0 .. qbase+15

    for (int kt = 0; kt < ntiles; ++kt) {
        const int kb = kt * 64;

        // ---- load 8 K B-fragments (one clause), then S = Q*K^T ----
        const unsigned short* kp =
            kbuf + ((size_t)b * T_ + kb + lrow) * HS_ + lhi * 16;
        v16bf bk[8];
        #pragma unroll
        for (int g = 0; g < 4; ++g) {
            bk[g * 2]     = *(const v16bf*)(kp + (size_t)g * 16 * HS_);
            bk[g * 2 + 1] = *(const v16bf*)(kp + (size_t)g * 16 * HS_ + 32);
        }
        v8f s[4];
        #pragma unroll
        for (int g = 0; g < 4; ++g) {
            s[g] = (v8f)(0.0f);
            s[g] = wmma_bf16(aq[0].v, bk[g * 2],     s[g]);
            s[g] = wmma_bf16(aq[1].v, bk[g * 2 + 1], s[g]);
        }

        // ---- issue V loads now: latency hides behind the softmax VALU ----
        const unsigned short* vp =
            vTbuf + ((size_t)b * HS_ + lrow) * T_ + kb + lhi * 16;
        v16bf bv[8];
        #pragma unroll
        for (int t = 0; t < 4; ++t) {
            bv[t * 2]     = *(const v16bf*)(vp + (size_t)t * 16 * T_);
            bv[t * 2 + 1] = *(const v16bf*)(vp + (size_t)t * 16 * T_ + 32);
        }

        if (kt + 1 < ntiles) {
            __builtin_prefetch(kp + (size_t)64 * HS_, 0, 0);  // global_prefetch_b8
        }

        const bool needMask = (kb + 63 > qbase);

        // ---- online softmax (C-layout: row = qbase+lhi*8+r, col g*16+lrow) ----
        #pragma unroll
        for (int r = 0; r < 8; ++r) {
            const int rowg = qbase + lhi * 8 + r;
            float sr[4];
            #pragma unroll
            for (int g = 0; g < 4; ++g) {
                sr[g] = s[g][r];
                if (needMask && (kb + g * 16 + lrow > rowg)) sr[g] = -1e30f;
            }
            float tm = fmaxf(fmaxf(sr[0], sr[1]), fmaxf(sr[2], sr[3]));
            tm = fmaxf(tm, __shfl_xor(tm, 1));
            tm = fmaxf(tm, __shfl_xor(tm, 2));
            tm = fmaxf(tm, __shfl_xor(tm, 4));
            tm = fmaxf(tm, __shfl_xor(tm, 8));     // reduce over 16-lane half
            float mn = fmaxf(m[r], tm);
            float al = __builtin_exp2f((m[r] - mn) * L2E);
            float p[4];
            #pragma unroll
            for (int g = 0; g < 4; ++g)
                p[g] = __builtin_exp2f((sr[g] - mn) * L2E);
            float rs = (p[0] + p[1]) + (p[2] + p[3]);
            rs += __shfl_xor(rs, 1);
            rs += __shfl_xor(rs, 2);
            rs += __shfl_xor(rs, 4);
            rs += __shfl_xor(rs, 8);
            lsum[r] = lsum[r] * al + rs;
            m[r] = mn;
            O[0][r] *= al; O[1][r] *= al; O[2][r] *= al; O[3][r] *= al;
            // stage P (bf16, row-major 16x64) for the C->A layout change
            const int prow = lhi * 8 + r;
            #pragma unroll
            for (int g = 0; g < 4; ++g)
                pLds[wave][prow * 64 + g * 16 + lrow] = f2bf(p[g]);
        }

        // ---- read P back in A-layout (same wave; LDS ops are in-order) ----
        AFrag ap[2];
        #pragma unroll
        for (int h = 0; h < 2; ++h) {
            ap[h].h[0] = *(const v8bf*)&pLds[wave][lrow * 64 + h * 32 + lhi * 16];
            ap[h].h[1] = *(const v8bf*)&pLds[wave][lrow * 64 + h * 32 + lhi * 16 + 8];
        }

        // ---- O += P * V ----
        #pragma unroll
        for (int t = 0; t < 4; ++t) {
            O[t] = wmma_bf16(ap[0].v, bv[t * 2],     O[t]);
            O[t] = wmma_bf16(ap[1].v, bv[t * 2 + 1], O[t]);
        }
    }

    // ---- epilogue: normalize and store fp32 [B,T,64] ----
    #pragma unroll
    for (int r = 0; r < 8; ++r) {
        float inv = 1.0f / lsum[r];
        int rowg = qbase + lhi * 8 + r;
        float* orow = out + ((size_t)b * T_ + rowg) * HS_;
        orow[0 * 16 + lrow] = O[0][r] * inv;
        orow[1 * 16 + lrow] = O[1][r] * inv;
        orow[2 * 16 + lrow] = O[2][r] * inv;
        orow[3 * 16 + lrow] = O[3][r] * inv;
    }
}

// ---------------------------------------------------------------------------
// Launch: ws layout = Wt(K|Q|V contiguous) | q | k | vT  (all bf16/u16)
//   3*64*1024*2 + 3*4*4096*64*2  ≈ 6.7 MB of workspace
// ---------------------------------------------------------------------------
extern "C" void kernel_launch(void* const* d_in, const int* in_sizes, int n_in,
                              void* d_out, int out_size, void* d_ws, size_t ws_size,
                              hipStream_t stream) {
    (void)in_sizes; (void)n_in; (void)out_size; (void)ws_size;
    const float* x  = (const float*)d_in[0];
    const float* Wk = (const float*)d_in[1];
    const float* Wq = (const float*)d_in[2];
    const float* Wv = (const float*)d_in[3];
    float* out = (float*)d_out;

    unsigned short* wt  = (unsigned short*)d_ws;           // 3*64*1024 contiguous
    unsigned short* qb  = wt + (size_t)3 * HS_ * C_;
    unsigned short* kb  = qb + (size_t)B_ * T_ * HS_;
    unsigned short* vtb = kb + (size_t)B_ * T_ * HS_;

    wt_convert_kernel<<<(HS_ * C_ + 255) / 256, 256, 0, stream>>>(
        Wk, Wq, Wv, wt);

    qkv_proj_kernel<<<dim3(T_ / 64, B_), 256, 0, stream>>>(
        x, wt, qb, kb, vtb);

    flash_attn_kernel<<<dim3(T_ / 64, B_), 128, 0, stream>>>(qb, kb, vtb, out);
}
